// MultiDiscriminator_72164040507566
// MI455X (gfx1250) — matrix operand order repaired and checked by reading, not compiled, not run on hardware
//
#include <hip/hip_runtime.h>
#include <math.h>

// ---------- types for CDNA5 WMMA ----------
typedef __attribute__((ext_vector_type(16))) __bf16 v16bf;
typedef __attribute__((ext_vector_type(8)))  float  v8f;

#define S_SKILLS 16
#define OBS_D 376
#define ACT_D 17
#define IN_D 393         // OBS+ACT
#define KPAD1 416        // IN padded to multiple of 32
#define KT1 13           // 416/32
#define KT2 8            // 256/32
#define HID 256
#define MTILE 64         // rows per GEMM block (4 x 16-row WMMA subtiles)

__device__ __forceinline__ unsigned short f2bf(float f) {
    unsigned u = __float_as_uint(f);
    u += 0x7FFFu + ((u >> 16) & 1u);   // round-to-nearest-even
    return (unsigned short)(u >> 16);
}
__device__ __forceinline__ float bf2f(unsigned short h) {
    return __uint_as_float(((unsigned)h) << 16);
}

// ---------- 1: init workspace control state ----------
__global__ void k_init(int* counts, int* cursors, int* tileSkill, int* order,
                       int pmax, int maxTiles) {
    int i = blockIdx.x * blockDim.x + threadIdx.x;
    if (i < pmax)     order[i] = -1;
    if (i < S_SKILLS) { counts[i] = 0; cursors[i] = 0; }
    if (i < maxTiles) tileSkill[i] = -1;
}

// ---------- 2: skill histogram ----------
__global__ void k_hist(const int* __restrict__ sk, int* counts, int n) {
    int i = blockIdx.x * blockDim.x + threadIdx.x;
    if (i < n) atomicAdd(&counts[sk[i]], 1);
}

// ---------- 3: serial scan -> 64-row padded offsets + macro-tile table ----------
__global__ void k_scan(const int* __restrict__ counts, int* padOff,
                       int* tileSkill, int* tileRow, int* numTiles) {
    if (blockIdx.x == 0 && threadIdx.x == 0) {
        int off = 0, t = 0;
        for (int s = 0; s < S_SKILLS; ++s) {
            padOff[s] = off;
            int nt = (counts[s] + MTILE - 1) / MTILE;
            for (int i = 0; i < nt; ++i) { tileSkill[t] = s; tileRow[t] = off + i * MTILE; ++t; }
            off += nt * MTILE;
        }
        padOff[S_SKILLS] = off;
        *numTiles = t;
    }
}

// ---------- 4: scatter samples into grouped order ----------
__global__ void k_scatter(const int* __restrict__ sk, const int* __restrict__ padOff,
                          int* cursors, int* order, int n) {
    int i = blockIdx.x * blockDim.x + threadIdx.x;
    if (i >= n) return;
    int s = sk[i];
    int pos = padOff[s] + atomicAdd(&cursors[s], 1);
    order[pos] = i;
}

// ---------- 5: convert + swizzle weights into WMMA B-fragment layout ----------
// dst index = (((s*KT + kt)*16 + nt)*512) + lane*16 + e
// lane 0-15 : N = nt*16 + lane,    K = kt*32 + e
// lane 16-31: N = nt*16 + lane-16, K = kt*32 + 16 + e
__global__ void k_convW(const float* __restrict__ W, unsigned short* __restrict__ Wd,
                        int KT, int Kreal, int total) {
    int idx = blockIdx.x * blockDim.x + threadIdx.x;
    if (idx >= total) return;
    int e    = idx & 15;
    int l    = (idx >> 4) & 31;
    int frag = idx >> 9;
    int nt   = frag & 15;
    int rest = frag >> 4;
    int kt   = rest % KT;
    int s    = rest / KT;
    int k = kt * 32 + (l >> 4) * 16 + e;
    int n = nt * 16 + (l & 15);
    float v = (k < Kreal) ? W[((size_t)s * Kreal + k) * HID + n] : 0.f;
    Wd[idx] = f2bf(v);
}

// ---------- 6: build grouped, padded bf16 input X = [obs || act] ----------
__global__ void k_buildX(const float* __restrict__ obs, const float* __restrict__ act,
                         const int* __restrict__ order, unsigned short* __restrict__ Xb) {
    int p = blockIdx.x;
    int i = order[p];
    for (int k = threadIdx.x; k < KPAD1; k += blockDim.x) {
        float v = 0.f;
        if (i >= 0 && k < IN_D)
            v = (k < OBS_D) ? obs[(size_t)i * OBS_D + k] : act[(size_t)i * ACT_D + (k - OBS_D)];
        Xb[(size_t)p * KPAD1 + k] = f2bf(v);
    }
}

// ---------- 7: grouped GEMM macro-tile kernel ----------
// Block = 8 waves = 64 rows x 256 cols. Wave w owns cols [w*32, w*32+32).
// A tile (64x32 bf16) staged through double-buffered LDS; weights read as
// pre-swizzled contiguous B fragments. 8 WMMAs per wave per K-step.
__global__ void __launch_bounds__(256)
k_gemm(const unsigned short* __restrict__ A, int lda, int ksteps,
       const unsigned short* __restrict__ Bw, int KT,
       const float* __restrict__ bias,
       const int* __restrict__ tileSkill, const int* __restrict__ tileRow,
       unsigned short* __restrict__ Out) {
    __shared__ __align__(16) unsigned short Ash[2][MTILE * 32];

    int t = blockIdx.x;
    int s = tileSkill[t];
    if (s < 0) return;                   // block-uniform exit (before barriers)
    int row0 = tileRow[t];
    int tid  = threadIdx.x;
    int wave = tid >> 5;
    int lane = tid & 31;
    int half = lane >> 4;

    // staging map: 64 rows x 4 chunks of 8 bf16 (one b128 per thread)
    int srow = tid >> 2;
    int schk = (tid & 3) * 8;
    const unsigned short* Ag = A + (size_t)(row0 + srow) * lda + schk;

    v8f c[4][2];
#pragma unroll
    for (int m = 0; m < 4; ++m) { c[m][0] = (v8f){}; c[m][1] = (v8f){}; }

    const unsigned short* Bsk = Bw + (size_t)s * KT * 16 * 512 + (size_t)lane * 16;
    int nt0 = wave * 2, nt1 = nt0 + 1;

    // prologue: stage K-step 0 into LDS buffer 0
    {
        uint4 st = *(const uint4*)Ag;
        *(uint4*)&Ash[0][srow * 32 + schk] = st;
    }
    __syncthreads();

    for (int kt = 0; kt < ksteps; ++kt) {
        int cur = kt & 1;

        // B fragments for this K-step: contiguous 32B per lane (2 x b128)
        const unsigned short* bp0 = Bsk + ((size_t)kt * 16 + nt0) * 512;
        const unsigned short* bp1 = Bsk + ((size_t)kt * 16 + nt1) * 512;
        union { v16bf v; uint4 q[2]; } b0, b1;
        b0.q[0] = ((const uint4*)bp0)[0]; b0.q[1] = ((const uint4*)bp0)[1];
        b1.q[0] = ((const uint4*)bp1)[0]; b1.q[1] = ((const uint4*)bp1)[1];

        // next K-step A chunk: global -> registers (overlaps with WMMAs below)
        bool more = (kt + 1 < ksteps);
        uint4 nst;
        if (more) {
            nst = *(const uint4*)(Ag + (size_t)(kt + 1) * 32);
            __builtin_prefetch(bp0 + 16 * 512, 0, 3);   // next weight fragment
        }

        // all 4 A fragments from LDS first (one dscnt wait), then 8 WMMAs
        union { v16bf v; uint4 q[2]; } a[4];
#pragma unroll
        for (int m = 0; m < 4; ++m) {
            const unsigned short* ap = &Ash[cur][((lane & 15) + 16 * m) * 32];
            a[m].q[0] = *(const uint4*)(ap + half * 16);
            a[m].q[1] = *(const uint4*)(ap + half * 16 + 8);
        }
#pragma unroll
        for (int m = 0; m < 4; ++m) {
            c[m][0] = __builtin_amdgcn_wmma_f32_16x16x32_bf16(false, a[m].v, false, b0.v,
                                                              (short)0, c[m][0], false, false);
            c[m][1] = __builtin_amdgcn_wmma_f32_16x16x32_bf16(false, a[m].v, false, b1.v,
                                                              (short)0, c[m][1], false, false);
        }

        if (more) *(uint4*)&Ash[cur ^ 1][srow * 32 + schk] = nst;
        __syncthreads();
    }

    // epilogue: bias + ReLU + bf16 store
    // C/D layout: lanes 0-15 -> N=lane, M=r ; lanes 16-31 -> N=lane-16, M=r+8
    int n0 = wave * 32 + (lane & 15);
    int n1 = n0 + 16;
    float bi0 = bias[s * HID + n0];
    float bi1 = bias[s * HID + n1];
    int mBase = half * 8;
#pragma unroll
    for (int m = 0; m < 4; ++m) {
#pragma unroll
        for (int r = 0; r < 8; ++r) {
            int row = row0 + 16 * m + mBase + r;
            Out[(size_t)row * HID + n0] = f2bf(fmaxf(c[m][0][r] + bi0, 0.f));
            Out[(size_t)row * HID + n1] = f2bf(fmaxf(c[m][1][r] + bi1, 0.f));
        }
    }
}

// ---------- 8: final 256-dot + bias + sigmoid, scatter back ----------
__global__ void k_out(const unsigned short* __restrict__ H2,
                      const int* __restrict__ order, const int* __restrict__ sk,
                      const float* __restrict__ W3, const float* __restrict__ b3,
                      float* __restrict__ out, int pmax) {
    int p = blockIdx.x * blockDim.x + threadIdx.x;
    if (p >= pmax) return;
    int i = order[p];
    if (i < 0) return;
    int s = sk[i];
    const unsigned short* h = H2 + (size_t)p * HID;
    const float* w = W3 + (size_t)s * HID;
    float acc = b3[s];
#pragma unroll 4
    for (int k = 0; k < HID; ++k) acc += bf2f(h[k]) * w[k];
    out[i] = 1.f / (1.f + __expf(-acc));
}

extern "C" void kernel_launch(void* const* d_in, const int* in_sizes, int n_in,
                              void* d_out, int out_size, void* d_ws, size_t ws_size,
                              hipStream_t stream) {
    const float* obs = (const float*)d_in[0];
    const float* act = (const float*)d_in[1];
    const int*   sk  = (const int*)  d_in[2];
    const float* W1  = (const float*)d_in[3];
    const float* b1  = (const float*)d_in[4];
    const float* W2  = (const float*)d_in[5];
    const float* b2  = (const float*)d_in[6];
    const float* W3  = (const float*)d_in[7];
    const float* b3  = (const float*)d_in[8];
    float* out = (float*)d_out;

    const int B = in_sizes[2];                           // batch from skill_idx
    const int PMAX = B + MTILE * S_SKILLS;               // worst-case padded rows
    const int MAXT = (B + MTILE - 1) / MTILE + S_SKILLS; // worst-case macro tiles

    // ---- workspace bump allocator (256B aligned regions) ----
    char* base = (char*)d_ws;
    size_t off = 0;
    auto alloc = [&](size_t bytes) -> char* {
        char* p = base + off;
        off = (off + bytes + 255) & ~(size_t)255;
        return p;
    };
    int* counts    = (int*)alloc(S_SKILLS * 4);
    int* cursors   = (int*)alloc(S_SKILLS * 4);
    int* padOff    = (int*)alloc((S_SKILLS + 1) * 4);
    int* numTiles  = (int*)alloc(4);
    int* tileSkill = (int*)alloc((size_t)MAXT * 4);
    int* tileRow   = (int*)alloc((size_t)MAXT * 4);
    int* order     = (int*)alloc((size_t)PMAX * 4);
    unsigned short* W1s = (unsigned short*)alloc((size_t)S_SKILLS * KT1 * 16 * 512 * 2);
    unsigned short* W2s = (unsigned short*)alloc((size_t)S_SKILLS * KT2 * 16 * 512 * 2);
    unsigned short* H1  = (unsigned short*)alloc((size_t)PMAX * HID * 2);
    unsigned short* Xb  = (unsigned short*)alloc((size_t)PMAX * KPAD1 * 2);
    unsigned short* H2  = Xb;   // reuse X buffer after GEMM1 (HID*2 <= KPAD1*2)

    const int T = 256;
    k_init<<<(PMAX + T - 1) / T, T, 0, stream>>>(counts, cursors, tileSkill, order, PMAX, MAXT);
    k_hist<<<(B + T - 1) / T, T, 0, stream>>>(sk, counts, B);
    k_scan<<<1, 32, 0, stream>>>(counts, padOff, tileSkill, tileRow, numTiles);
    k_scatter<<<(B + T - 1) / T, T, 0, stream>>>(sk, padOff, cursors, order, B);

    int n1 = S_SKILLS * KT1 * 16 * 512;
    int n2 = S_SKILLS * KT2 * 16 * 512;
    k_convW<<<(n1 + T - 1) / T, T, 0, stream>>>(W1, W1s, KT1, IN_D, n1);
    k_convW<<<(n2 + T - 1) / T, T, 0, stream>>>(W2, W2s, KT2, HID, n2);

    k_buildX<<<PMAX, T, 0, stream>>>(obs, act, order, Xb);

    k_gemm<<<MAXT, T, 0, stream>>>(Xb, KPAD1, KT1, W1s, KT1, b1, tileSkill, tileRow, H1);
    k_gemm<<<MAXT, T, 0, stream>>>(H1, HID,  KT2, W2s, KT2, b2, tileSkill, tileRow, H2);

    k_out<<<(PMAX + T - 1) / T, T, 0, stream>>>(H2, order, sk, W3, b3, out, PMAX);
}
